// GaussianNonLocalBlock_36103495090205
// MI455X (gfx1250) — matrix operand order repaired and checked by reading, not compiled, not run on hardware
//
#include <hip/hip_runtime.h>
#include <hip/hip_bf16.h>

// Problem constants (from the reference)
#define BB 8
#define CC 256
#define EE 128
#define NN 3136          // 56*56
#define KTILES (NN / 64) // 49 key tiles of 64

typedef __attribute__((ext_vector_type(16))) __bf16 v16bf;
typedef __attribute__((ext_vector_type(8)))  __bf16 v8bf;
typedef __attribute__((ext_vector_type(8)))  float  v8f;

static __device__ __forceinline__ v8f vzero8() {
  v8f z = {0.f, 0.f, 0.f, 0.f, 0.f, 0.f, 0.f, 0.f};
  return z;
}

// A-matrix fragment (16x32 bf16). Source is row-major [m][k], row stride ld.
// ISA 7.12.2: lanes 0-15 hold row M=lane, K={0..7,16..23}; lanes 16-31 row
// M=lane-16, K={8..15,24..31}.  Two 16B loads per lane.
static __device__ __forceinline__ v16bf frag_a(const __bf16* base, int ld, int k0, int lane) {
  int m  = lane & 15;
  int kb = (lane >> 4) << 3; // 0 or 8
  const __bf16* p = base + (size_t)m * ld + (k0 + kb);
  v8bf lo = *(const v8bf*)(p);
  v8bf hi = *(const v8bf*)(p + 16);
  return __builtin_shufflevector(lo, hi, 0, 1, 2, 3, 4, 5, 6, 7,
                                 8, 9, 10, 11, 12, 13, 14, 15);
}

// B-matrix fragment (32x16 bf16). Element (k,n) = src[n*ld + k] ("n-major"):
// lanes 0-15 hold column N=lane, K={0..7,16..23}; lanes 16-31 column N=lane-16,
// K={8..15,24..31}.  Two 16B reads per lane.
static __device__ __forceinline__ v16bf frag_b(const __bf16* base, size_t ld, int k0, int lane) {
  int n  = lane & 15;
  int kb = (lane >> 4) << 3; // 0 or 8
  const __bf16* p = base + (size_t)n * ld + (k0 + kb);
  v8bf lo = *(const v8bf*)(p);
  v8bf hi = *(const v8bf*)(p + 16);
  return __builtin_shufflevector(lo, hi, 0, 1, 2, 3, 4, 5, 6, 7,
                                 8, 9, 10, 11, 12, 13, 14, 15);
}

// ---------------------------------------------------------------------------
// Kernel 1a: coalesced LDS tile transpose x (b,c,n) f32 -> xb (b,n,c) bf16.
// 64x64 tiles; LDS pitch 66 (33-word column stride -> bank-conflict-free).
// ---------------------------------------------------------------------------
__global__ __launch_bounds__(256) void k_transpose(const float* __restrict__ x,
                                                   __bf16* __restrict__ xb) {
  __shared__ __bf16 tile[64][66];
  int blk = blockIdx.x;
  int ct = blk % (CC / 64); blk /= (CC / 64);
  int nt = blk % (NN / 64);
  int b  = blk / (NN / 64);
  int c0 = ct * 64, n0 = nt * 64;
  int j  = threadIdx.x & 63; // n (read phase) / c (write phase)
  int i0 = threadIdx.x >> 6; // 0..3

  const float* xs = x + ((size_t)b * CC + c0) * NN + n0;
  #pragma unroll
  for (int i = i0; i < 64; i += 4)
    tile[i][j] = (__bf16)xs[(size_t)i * NN + j]; // coalesced 256B row reads
  __syncthreads();
  __bf16* xd = xb + ((size_t)b * NN + n0) * CC + c0;
  #pragma unroll
  for (int jj = i0; jj < 64; jj += 4)
    xd[(size_t)jj * CC + j] = tile[j][jj];       // coalesced bf16 row writes
}

// Kernel 1b: cast the 1x1-conv weights to bf16 (E*C == C*E).
__global__ __launch_bounds__(256) void k_castw(const float* __restrict__ vw,
                                               const float* __restrict__ ew,
                                               __bf16* __restrict__ vwb,
                                               __bf16* __restrict__ ewb) {
  int idx = blockIdx.x * 256 + threadIdx.x;
  if (idx < EE * CC) {
    vwb[idx] = (__bf16)vw[idx];
    ewb[idx] = (__bf16)ew[idx];
  }
}

// ---------------------------------------------------------------------------
// Kernel 2: vt[b][e][n] = sum_c v_w[e][c]*x[b][c][n] + v_b[e]   (WMMA)
// Stored e-major so the P@V B-fragments are contiguous per lane.
// Block = 4 waves; wave w computes one 16(e) x 16(n) tile.
// ---------------------------------------------------------------------------
__global__ __launch_bounds__(128) void k_vt(const __bf16* __restrict__ xb,
                                            const __bf16* __restrict__ vwb,
                                            const float* __restrict__ vb,
                                            __bf16* __restrict__ vt) {
  int lane = threadIdx.x & 31, wave = threadIdx.x >> 5;
  int half = lane >> 4, lan = lane & 15;
  int blk = blockIdx.x;
  int nch = blk % (NN / 64); blk /= (NN / 64);
  int et  = blk % (EE / 16);
  int b   = blk / (EE / 16);
  int n0 = nch * 64 + wave * 16;
  int e0 = et * 16;
  const __bf16* xbb = xb + (size_t)b * NN * CC;

  v8f acc;
  #pragma unroll
  for (int r = 0; r < 8; ++r) acc[r] = vb[e0 + r + 8 * half]; // bias as C init

  #pragma unroll
  for (int k0 = 0; k0 < CC; k0 += 32) {
    v16bf a  = frag_a(vwb + (size_t)e0 * CC, CC, k0, lane); // rows of v_w
    v16bf bm = frag_b(xbb + (size_t)n0 * CC, CC, k0, lane); // xb rows = cols of X
    acc = __builtin_amdgcn_wmma_f32_16x16x32_bf16(false, a, false, bm,
                                                  (short)0, acc, false, false);
  }
  int n = n0 + lan;
  #pragma unroll
  for (int r = 0; r < 8; ++r) {
    int e = e0 + r + 8 * half;
    vt[((size_t)b * EE + e) * NN + n] = (__bf16)acc[r];
  }
}

// ---------------------------------------------------------------------------
// Kernel 3: fused flash attention + output projection + bias + residual.
// Block = 4 waves, each wave owns 16 query rows of a 64-row tile.
// Q fragments are hoisted into registers for the whole key loop.
// ---------------------------------------------------------------------------
__global__ __launch_bounds__(128) void k_attn(const float* __restrict__ x,
                                              const __bf16* __restrict__ xb,
                                              const __bf16* __restrict__ vt,
                                              const __bf16* __restrict__ ewb,
                                              const float* __restrict__ eb,
                                              float* __restrict__ out) {
  __shared__ __bf16 stage[4][16 * 128]; // 16KB; wave-private staging (P / embed)
  int lane = threadIdx.x & 31, wave = threadIdx.x >> 5;
  int half = lane >> 4, lan = lane & 15;
  int b  = blockIdx.x / KTILES;
  int qt = blockIdx.x % KTILES;
  int q0 = qt * 64 + wave * 16; // this wave's 16 query rows
  const __bf16* xbb = xb + (size_t)b * NN * CC;
  const __bf16* vtb = vt + (size_t)b * EE * NN;
  __bf16* pst = &stage[wave][0];

  // Q tile (16 x 256) lives in registers across the whole key loop.
  v16bf qf[8];
  #pragma unroll
  for (int i = 0; i < 8; ++i)
    qf[i] = frag_a(xbb + (size_t)q0 * CC, CC, 32 * i, lane);

  v8f o[8]; // 16 x 128 f32 output accumulator (8 e-subtiles)
  #pragma unroll
  for (int je = 0; je < 8; ++je) o[je] = vzero8();
  float mrun[8], lrun[8];
  #pragma unroll
  for (int r = 0; r < 8; ++r) { mrun[r] = -3.0e38f; lrun[r] = 0.f; }

  for (int kt = 0; kt < KTILES; ++kt) {
    int n0k = kt * 64;
    if (kt + 1 < KTILES) // pull next key tile toward L2/L0
      __builtin_prefetch(xbb + (size_t)(n0k + 64) * CC, 0, 1);

    // ---- S = Q @ K^T  (16 x 64, K-dim = C = 256) ----
    v8f s[4];
    #pragma unroll
    for (int j = 0; j < 4; ++j) s[j] = vzero8();
    #pragma unroll
    for (int i = 0; i < 8; ++i) {
      #pragma unroll
      for (int j = 0; j < 4; ++j) {
        v16bf bm = frag_b(xbb + (size_t)(n0k + 16 * j) * CC, CC, 32 * i, lane);
        s[j] = __builtin_amdgcn_wmma_f32_16x16x32_bf16(false, qf[i], false, bm,
                                                       (short)0, s[j], false, false);
      }
    }

    // ---- online softmax (row = VGPR index within 16-lane half) ----
    #pragma unroll
    for (int r = 0; r < 8; ++r) {
      float t = fmaxf(fmaxf(s[0][r], s[1][r]), fmaxf(s[2][r], s[3][r]));
      #pragma unroll
      for (int msk = 1; msk < 16; msk <<= 1)
        t = fmaxf(t, __shfl_xor(t, msk, 32));
      float mn = fmaxf(mrun[r], t);
      float sc = __expf(mrun[r] - mn);
      mrun[r] = mn;
      float rs = 0.f;
      #pragma unroll
      for (int j = 0; j < 4; ++j) {
        float p = __expf(s[j][r] - mn);
        s[j][r] = p;
        rs += p;
      }
      #pragma unroll
      for (int msk = 1; msk < 16; msk <<= 1)
        rs += __shfl_xor(rs, msk, 32);
      lrun[r] = lrun[r] * sc + rs;
      #pragma unroll
      for (int je = 0; je < 8; ++je) o[je][r] *= sc;
      // stash P (C/D layout -> row-major bf16 in LDS, ld = 64)
      #pragma unroll
      for (int j = 0; j < 4; ++j)
        pst[(r + 8 * half) * 64 + 16 * j + lan] = (__bf16)s[j][r];
    }

    // ---- O += P @ V  (same-wave LDS ops are in-order; no barrier) ----
    #pragma unroll
    for (int kk = 0; kk < 64; kk += 32) {
      v16bf a = frag_a(pst, 64, kk, lane);
      #pragma unroll
      for (int je = 0; je < 8; ++je) {
        v16bf bm = frag_b(vtb + (size_t)(je * 16) * NN, NN, n0k + kk, lane);
        o[je] = __builtin_amdgcn_wmma_f32_16x16x32_bf16(false, a, false, bm,
                                                        (short)0, o[je], false, false);
      }
    }
  }

  // ---- normalize, stage embed (16 x 128, ld = 128) ----
  #pragma unroll
  for (int r = 0; r < 8; ++r) {
    float inv = 1.f / lrun[r];
    #pragma unroll
    for (int je = 0; je < 8; ++je)
      pst[(r + 8 * half) * 128 + je * 16 + lan] = (__bf16)(o[je][r] * inv);
  }

  // ---- out[c][n] = embed @ e_w^T + e_b[c] + x[c][n] ----
  for (int jc = 0; jc < 16; ++jc) {
    v8f d = vzero8();
    #pragma unroll
    for (int k0 = 0; k0 < EE; k0 += 32) {
      v16bf a  = frag_a(pst, 128, k0, lane);
      v16bf bm = frag_b(ewb + (size_t)(jc * 16) * EE, EE, k0, lane);
      d = __builtin_amdgcn_wmma_f32_16x16x32_bf16(false, a, false, bm,
                                                  (short)0, d, false, false);
    }
    int c = jc * 16 + lan;
    float ebv = eb[c];
    size_t rowbase = ((size_t)b * CC + c) * NN + q0;
    #pragma unroll
    for (int r = 0; r < 8; ++r) {
      size_t idx = rowbase + r + 8 * half;
      out[idx] = d[r] + x[idx] + ebv;
    }
  }
}

// ---------------------------------------------------------------------------
extern "C" void kernel_launch(void* const* d_in, const int* in_sizes, int n_in,
                              void* d_out, int out_size, void* d_ws, size_t ws_size,
                              hipStream_t stream) {
  (void)in_sizes; (void)n_in; (void)out_size; (void)ws_size;
  const float* x  = (const float*)d_in[0];
  const float* vw = (const float*)d_in[1];
  const float* vb = (const float*)d_in[2];
  const float* ew = (const float*)d_in[3];
  const float* eb = (const float*)d_in[4];
  float* out = (float*)d_out;

  // Workspace layout (bf16): xb (B,N,C) | vt (B,E,N) | vwb (E,C) | ewb (C,E)
  // = 12.8MB + 6.4MB + 64KB + 64KB ~= 19.4MB
  __bf16* xb  = (__bf16*)d_ws;
  __bf16* vt  = xb  + (size_t)BB * NN * CC;
  __bf16* vwb = vt  + (size_t)BB * EE * NN;
  __bf16* ewb = vwb + (size_t)EE * CC;

  {
    int grid = BB * (NN / 64) * (CC / 64); // 1568 tiles
    k_transpose<<<grid, 256, 0, stream>>>(x, xb);
  }
  {
    k_castw<<<(EE * CC + 255) / 256, 256, 0, stream>>>(vw, ew, vwb, ewb);
  }
  {
    int grid = BB * (EE / 16) * (NN / 64); // 3136 blocks
    k_vt<<<grid, 128, 0, stream>>>(xb, vwb, vb, vt);
  }
  {
    int grid = BB * KTILES; // 392 blocks, 4 waves each
    k_attn<<<grid, 128, 0, stream>>>(x, xb, vt, ewb, eb, out);
  }
}
